// MultiHeadAttention_40200893890578
// MI455X (gfx1250) — matrix operand order repaired and checked by reading, not compiled, not run on hardware
//
#include <hip/hip_runtime.h>
#include <hip/hip_bf16.h>
#include <cstdint>

typedef _Float16 v16h __attribute__((ext_vector_type(16)));
typedef _Float16 v8h  __attribute__((ext_vector_type(8)));
typedef float    v8f  __attribute__((ext_vector_type(8)));

#define NHEADS 8
#define DHEAD  64
#define DMODEL 512
#define BATCH  16
#define NQ     1024
#define NK     1024
#define NORMS  0.125f           // 1/sqrt(64)
#define NEG_BIG -1e30f

#if defined(__has_builtin)
#if __has_builtin(__builtin_amdgcn_tensor_load_to_lds) && __has_builtin(__builtin_amdgcn_s_wait_tensorcnt)
#define USE_TDM 1
#endif
#endif

#ifdef USE_TDM
typedef unsigned int v4u __attribute__((ext_vector_type(4)));
typedef int          v8i __attribute__((ext_vector_type(8)));
typedef int          v4i __attribute__((ext_vector_type(4)));
#endif

// ---------------------------------------------------------------------------
// WMMA wrapper: D = A(16x32 f16) * B(32x16 f16) + C(16x16 f32)
// ---------------------------------------------------------------------------
__device__ __forceinline__ v8f wmma_f16(v16h a, v16h b, v8f c) {
  return __builtin_amdgcn_wmma_f32_16x16x32_f16(
      /*neg_a=*/false, a, /*neg_b=*/false, b,
      /*c_mod=*/(short)0, c, /*reuse_a=*/false, /*reuse_b=*/false);
}

// A-fragment (16x32, MxK) from a row-major f32 matrix, converting to f16.
// Lanes 0-15 (row M=lane) hold K{0..7,16..23}; lanes 16-31 hold K{8..15,24..31}.
__device__ __forceinline__ v16h a_frag_f32(const float* mat, long ld, int m0, int k0) {
  int lane = threadIdx.x & 31;
  int r = lane & 15, hi = lane >> 4;
  const float* p = mat + (long)(m0 + r) * ld + k0 + hi * 8;
  v8f x = *(const v8f*)p;          // K = k0 + hi*8 + (0..7)
  v8f y = *(const v8f*)(p + 16);   // K = k0 + 16 + hi*8 + (0..7)
  v16h a;
#pragma unroll
  for (int i = 0; i < 8; ++i) { a[i] = (_Float16)x[i]; a[i + 8] = (_Float16)y[i]; }
  return a;
}

// A-fragment (16x32) from a row-major f16 matrix.
__device__ __forceinline__ v16h a_frag_f16(const _Float16* mat, long ld, int m0, int k0) {
  int lane = threadIdx.x & 31;
  int r = lane & 15, hi = lane >> 4;
  const _Float16* p = mat + (long)(m0 + r) * ld + k0 + hi * 8;
  v8h x = *(const v8h*)p;
  v8h y = *(const v8h*)(p + 16);
  v16h a;
#pragma unroll
  for (int i = 0; i < 8; ++i) { a[i] = x[i]; a[i + 8] = y[i]; }
  return a;
}

// B-fragment (32x16, KxN) where each column's K elements are contiguous:
// column n base = mat + n*ldcol. Lanes 0-15 (N=lane) hold K{0..7,16..23}, lanes
// 16-31 hold K{8..15,24..31} (two 16B runs per lane).
__device__ __forceinline__ v16h b_frag_kcontig(const _Float16* mat, long ldcol) {
  int lane = threadIdx.x & 31;
  int n = lane & 15, hi = lane >> 4;
  const _Float16* p = mat + (long)n * ldcol + hi * 8;
  v8h x = *(const v8h*)p;
  v8h y = *(const v8h*)(p + 16);
  v16h b;
#pragma unroll
  for (int i = 0; i < 8; ++i) { b[i] = x[i]; b[i + 8] = y[i]; }
  return b;
}

// ---------------------------------------------------------------------------
// Kernel 1: fused Q/K/V projection.  X(16384x512 f32) * W(512x512) -> f16.
// Block = 256 threads (8 waves), tile M=128 x N=64 (one full head strip).
// Each wave: 16 rows x 64 cols = 4 C-tiles sharing one A-fragment -> 4 WMMA
// per A-load, and X re-read factor drops to 512/64 = 8 (all L2-resident).
// ---------------------------------------------------------------------------
__global__ __launch_bounds__(256)
void mha_proj_kernel(const float* __restrict__ q, const float* __restrict__ k,
                     const float* __restrict__ v,
                     const float* __restrict__ Wq, const float* __restrict__ Wk,
                     const float* __restrict__ Wv,
                     _Float16* __restrict__ Qh, _Float16* __restrict__ Kh,
                     _Float16* __restrict__ Vh) {
  __shared__ __align__(16) _Float16 wt[64 * 40];   // W^T tile: [n][k], ld=40
  int tid = threadIdx.x;
  int wave = tid >> 5;
  int mat = blockIdx.z;
  const float* X = (mat == 0) ? q : (mat == 1 ? k : v);
  const float* W = (mat == 0) ? Wq : (mat == 1 ? Wk : Wv);
  _Float16*    O = (mat == 0) ? Qh : (mat == 1 ? Kh : Vh);

  int m0 = blockIdx.x * 128 + wave * 16;
  int n0 = blockIdx.y * 64;          // one head per N-block
  int h  = blockIdx.y;

  v8f acc[4] = {{}, {}, {}, {}};
  for (int k0 = 0; k0 < DMODEL; k0 += 32) {
    __syncthreads();
    // Stage transposed weight tile (64 n x 32 k), converting f32->f16.
#pragma unroll
    for (int i = 0; i < 8; ++i) {
      int idx = tid + i * 256;          // 2048 elements total
      int nn = idx >> 5, kk = idx & 31;
      wt[nn * 40 + kk] =
          (_Float16)W[(long)h * DMODEL * DHEAD + (long)(k0 + kk) * DHEAD + nn];
    }
    __syncthreads();
    v16h a = a_frag_f32(X, DMODEL, m0, k0);
#pragma unroll
    for (int t = 0; t < 4; ++t)
      acc[t] = wmma_f16(a, b_frag_kcontig(wt + t * 16 * 40, 40), acc[t]);
  }
  // C layout: N = lane&15, M = 8*(lane>>4) + vgpr
  int lane = tid & 31;
  int nn = lane & 15, hi = lane >> 4;
#pragma unroll
  for (int t = 0; t < 4; ++t)
#pragma unroll
    for (int vv = 0; vv < 8; ++vv)
      O[(long)(m0 + hi * 8 + vv) * DMODEL + n0 + t * 16 + nn] = (_Float16)acc[t][vv];
}

// ---------------------------------------------------------------------------
// Kernel 2: flash attention. One block (8 waves, 256 thr) per (b, h, 128-q
// strip); each wave owns 16 queries. Key tiles of 32, online softmax.
// K tile is staged into LDS by the Tensor Data Mover (wave 0 issues one
// tensor_load_to_lds per tile, waits on TENSORcnt, barrier publishes).
// ---------------------------------------------------------------------------
__global__ __launch_bounds__(256)
void mha_flash_kernel(const _Float16* __restrict__ Qh,
                      const _Float16* __restrict__ Kh,
                      const _Float16* __restrict__ Vh,
                      const unsigned char* __restrict__ mask,
                      _Float16* __restrict__ Hh) {
  __shared__ __align__(16) _Float16 kt[32 * 64];        // K tile: [key][c], ld=64
  __shared__ __align__(16) _Float16 vt[64 * 40];        // V^T tile: [c][key]
  __shared__ __align__(16) _Float16 pbuf[8][16 * 40];   // per-wave P: [row][key]

  int tid = threadIdx.x, wave = tid >> 5, lane = tid & 31;
  int n = lane & 15, hi = lane >> 4;
  int b = blockIdx.y, h = blockIdx.z;
  int q0 = blockIdx.x * 128 + wave * 16;

  const _Float16* Qb = Qh + (long)b * NQ * DMODEL + h * DHEAD;
  const _Float16* Kb = Kh + (long)b * NK * DMODEL + h * DHEAD;
  const _Float16* Vb = Vh + (long)b * NK * DMODEL + h * DHEAD;
  const unsigned char* Mb = mask + (long)b * NQ * NK;

  // Q fragments for this wave's 16 queries (gemm-K = 64 channels, 2 chunks)
  v16h qa0 = a_frag_f16(Qb, DMODEL, q0, 0);
  v16h qa1 = a_frag_f16(Qb, DMODEL, q0, 32);

  float mrow[8], lrow[8];
  v8f o0 = {}, o1 = {}, o2 = {}, o3 = {};
#pragma unroll
  for (int i = 0; i < 8; ++i) { mrow[i] = NEG_BIG; lrow[i] = 0.f; }

  _Float16* pw = &pbuf[wave][0];

  for (int g0 = 0; g0 < NK; g0 += 32) {
    __syncthreads();   // previous-iteration consumers of kt/vt are done

    // --- Stage K tile (32 keys x 64 ch, no transpose needed) --------------
#ifdef USE_TDM
    if (wave == 0) {
      // Tensor DMA descriptor (ISA 8.3/8.4): 2D tile 64 x 32, f16, stride 512.
      uint64_t ga  = (uint64_t)(uintptr_t)(Kb + (long)g0 * DMODEL);
      uint32_t lds = (uint32_t)(uintptr_t)&kt[0];   // low 32 bits = LDS offset
      v4u d0;
      d0[0] = 1u;                                   // count=1 (valid), user mode
      d0[1] = lds;                                  // lds_addr
      d0[2] = (uint32_t)ga;                         // global_addr[31:0]
      d0[3] = (uint32_t)((ga >> 32) & 0x01FFFFFFu) | (2u << 30);  // [56:32]|type=2
      v8i d1;
      d1[0] = (int)(1u << 16);                      // data_size=1 (2 bytes)
      d1[1] = (int)(64u << 16);                     // tensor_dim0[15:0]=64
      d1[2] = (int)(32u << 16);                     // dim0 hi=0 | tensor_dim1[15:0]=32
      d1[3] = (int)(64u << 16);                     // dim1 hi=0 | tile_dim0=64
      d1[4] = (int)32u;                             // tile_dim1=32 | tile_dim2=0
      d1[5] = (int)DMODEL;                          // tensor_dim0_stride = 512
      d1[6] = 0;
      d1[7] = 0;
      v4i dz4 = {};
      v8i dz8 = {};
      // clang-23 / therock 6-arg form: (g0, g1, g2, g3, g4, cpol)
      __builtin_amdgcn_tensor_load_to_lds(d0, d1, dz4, dz4, dz8, 0);
      __builtin_amdgcn_s_wait_tensorcnt(0);
    }
#else
    {
      int key = tid >> 3;
      int cb  = (tid & 7) * 8;
      *(v8h*)(kt + key * 64 + cb) = *(const v8h*)(Kb + (long)(g0 + key) * DMODEL + cb);
    }
#endif

    // --- Stage V tile transposed: vt[c][key] (2048 f16, 8 per thread) -----
    {
      int key = tid >> 3;
      int cb  = (tid & 7) * 8;
      v8h vv = *(const v8h*)(Vb + (long)(g0 + key) * DMODEL + cb);
#pragma unroll
      for (int i = 0; i < 8; ++i) vt[(cb + i) * 40 + key] = vv[i];
    }
    __syncthreads();   // kt (TDM) + vt visible to all waves

    if (g0 + 32 < NK) {  // global_prefetch_b8 of next K/V tiles into GL2
      __builtin_prefetch(Kb + (long)(g0 + 32 + n) * DMODEL, 0, 1);
      __builtin_prefetch(Vb + (long)(g0 + 32 + n) * DMODEL, 0, 1);
    }

    // S(16x32) = Q(16x64) * K^T : 2 n-tiles x 2 k-chunks, B frags from LDS.
    v8f s0 = {}, s1 = {};
    s0 = wmma_f16(qa0, b_frag_kcontig(kt, 64), s0);
    s0 = wmma_f16(qa1, b_frag_kcontig(kt + 32, 64), s0);
    s1 = wmma_f16(qa0, b_frag_kcontig(kt + 16 * 64, 64), s1);
    s1 = wmma_f16(qa1, b_frag_kcontig(kt + 16 * 64 + 32, 64), s1);

    // Mask + online softmax. Each lane carries 8 rows (M = 8*hi + vv).
#pragma unroll
    for (int vv = 0; vv < 8; ++vv) {
      int qrow = q0 + hi * 8 + vv;
      const unsigned char* mp = Mb + (long)qrow * NK + g0;
      float sa = (mp[n]      != 0) ? NEG_BIG : s0[vv] * NORMS;
      float sb = (mp[16 + n] != 0) ? NEG_BIG : s1[vv] * NORMS;
      float t = fmaxf(sa, sb);                    // row lives in one 16-lane half
      t = fmaxf(t, __shfl_xor(t, 1, 32));
      t = fmaxf(t, __shfl_xor(t, 2, 32));
      t = fmaxf(t, __shfl_xor(t, 4, 32));
      t = fmaxf(t, __shfl_xor(t, 8, 32));
      float mnew  = fmaxf(mrow[vv], t);
      float alpha = __expf(mrow[vv] - mnew);
      float pa = (sa == NEG_BIG) ? 0.f : __expf(sa - mnew);
      float pb = (sb == NEG_BIG) ? 0.f : __expf(sb - mnew);
      float ps = pa + pb;
      ps += __shfl_xor(ps, 1, 32);
      ps += __shfl_xor(ps, 2, 32);
      ps += __shfl_xor(ps, 4, 32);
      ps += __shfl_xor(ps, 8, 32);
      lrow[vv] = lrow[vv] * alpha + ps;
      mrow[vv] = mnew;
      o0[vv] *= alpha; o1[vv] *= alpha; o2[vv] *= alpha; o3[vv] *= alpha;
      // C-layout -> LDS so we can reload P in A-layout
      pw[(hi * 8 + vv) * 40 + n]      = (_Float16)pa;
      pw[(hi * 8 + vv) * 40 + 16 + n] = (_Float16)pb;
    }
    __syncthreads();   // P ordering + protect kt/vt reuse

    // O(16x64) += P(16x32) * V(32x64): 4 n-tiles of 16 channels
    v16h pf = a_frag_f16(pw, 40, 0, 0);
    o0 = wmma_f16(pf, b_frag_kcontig(vt +  0 * 40, 40), o0);
    o1 = wmma_f16(pf, b_frag_kcontig(vt + 16 * 40, 40), o1);
    o2 = wmma_f16(pf, b_frag_kcontig(vt + 32 * 40, 40), o2);
    o3 = wmma_f16(pf, b_frag_kcontig(vt + 48 * 40, 40), o3);
  }

  // Normalize and emit heads in [b][q][h*64+c] layout (f16).
  _Float16* Ob = Hh + (long)b * NQ * DMODEL + h * DHEAD;
#pragma unroll
  for (int vv = 0; vv < 8; ++vv) {
    float inv = (lrow[vv] > 0.f) ? 1.f / lrow[vv] : 0.f;
    long row = (long)(q0 + hi * 8 + vv) * DMODEL;
    Ob[row +  0 + n] = (_Float16)(o0[vv] * inv);
    Ob[row + 16 + n] = (_Float16)(o1[vv] * inv);
    Ob[row + 32 + n] = (_Float16)(o2[vv] * inv);
    Ob[row + 48 + n] = (_Float16)(o3[vv] * inv);
  }
}

// ---------------------------------------------------------------------------
// Kernel 3: output projection. heads(16384x512 f16) * W_out(512x512) -> f32.
// W_out[h][v][e] flattens exactly to a row-major 512x512 matrix.
// Same 128x64 blocking as kernel 1 (4 WMMA per A-fragment).
// ---------------------------------------------------------------------------
__global__ __launch_bounds__(256)
void mha_outproj_kernel(const _Float16* __restrict__ Hh,
                        const float* __restrict__ Wo,
                        float* __restrict__ out) {
  __shared__ __align__(16) _Float16 wt[64 * 40];
  int tid = threadIdx.x, wave = tid >> 5, lane = tid & 31;
  int m0 = blockIdx.x * 128 + wave * 16;
  int n0 = blockIdx.y * 64;

  v8f acc[4] = {{}, {}, {}, {}};
  for (int k0 = 0; k0 < DMODEL; k0 += 32) {
    __syncthreads();
#pragma unroll
    for (int i = 0; i < 8; ++i) {
      int idx = tid + i * 256;
      int nn = idx >> 5, kk = idx & 31;
      wt[nn * 40 + kk] = (_Float16)Wo[(long)(k0 + kk) * DMODEL + n0 + nn];
    }
    __syncthreads();
    v16h a = a_frag_f16(Hh, DMODEL, m0, k0);
#pragma unroll
    for (int t = 0; t < 4; ++t)
      acc[t] = wmma_f16(a, b_frag_kcontig(wt + t * 16 * 40, 40), acc[t]);
  }
  int nn = lane & 15, hi = lane >> 4;
#pragma unroll
  for (int t = 0; t < 4; ++t)
#pragma unroll
    for (int vv = 0; vv < 8; ++vv)
      out[(long)(m0 + hi * 8 + vv) * DMODEL + n0 + t * 16 + nn] = acc[t][vv];
}

// ---------------------------------------------------------------------------
extern "C" void kernel_launch(void* const* d_in, const int* in_sizes, int n_in,
                              void* d_out, int out_size, void* d_ws, size_t ws_size,
                              hipStream_t stream) {
  const float* q  = (const float*)d_in[0];
  const float* k  = (const float*)d_in[1];
  const float* v  = (const float*)d_in[2];
  const unsigned char* mask = (const unsigned char*)d_in[3];  // jnp.bool_ = 1 byte
  const float* Wq = (const float*)d_in[4];
  const float* Wk = (const float*)d_in[5];
  const float* Wv = (const float*)d_in[6];
  const float* Wo = (const float*)d_in[7];
  float* out = (float*)d_out;

  // Workspace: 4 x (16384 x 512) f16 tensors = 64 MB, fully overwritten.
  const long T = (long)BATCH * NQ * DMODEL;   // 8,388,608 elements
  _Float16* ws = (_Float16*)d_ws;
  _Float16* Qh = ws;
  _Float16* Kh = ws + T;
  _Float16* Vh = ws + 2 * T;
  _Float16* Hh = ws + 3 * T;

  // 1) Q/K/V projections: grid = (M/128, N/64, {q,k,v})
  mha_proj_kernel<<<dim3(128, 8, 3), 256, 0, stream>>>(q, k, v, Wq, Wk, Wv, Qh, Kh, Vh);
  // 2) flash attention: grid = (NQ/128, B, H)
  mha_flash_kernel<<<dim3(8, 16, 8), 256, 0, stream>>>(Qh, Kh, Vh, mask, Hh);
  // 3) output projection
  mha_outproj_kernel<<<dim3(128, 8), 256, 0, stream>>>(Hh, Wo, out);
}
